// Attention_6562710028918
// MI455X (gfx1250) — matrix operand order repaired and checked by reading, not compiled, not run on hardware
//
#include <hip/hip_runtime.h>

// =====================================================================
// CIM attention for MI455X (gfx1250, wave32, WMMA 16x16x32 f16->f32)
//   B=8, N=1024, C=512, H=4, HD=128
// Pipeline (f16 data, f32 accumulation):
//   1) cvt: x, w_qkv (SCALE folded into Q rows), w_proj -> f16 in ws
//   2) qkv GEMM  [8192x512] x [1536x512]^T -> Q/K/V [B,H,N,HD] f16
//      (async-to-LDS staging, double-buffered)
//   3) fused flash-style attention with cross-head mix (CIM) -> AO f16
//      (async-to-LDS K tiles overlapped with V-transpose staging)
//   4) proj GEMM [8192x512] x [512x512]^T + bias -> d_out f32
// Workspace use: ~44 MB.
// =====================================================================

constexpr int BB = 8, NN = 1024, CC = 512, HH = 4, HD = 128;
constexpr float QK_SCALE = 0.08838834764831845f;   // 128^-0.5

typedef __attribute__((ext_vector_type(16))) _Float16 v16h;
typedef __attribute__((ext_vector_type(8)))  _Float16 v8h;
typedef __attribute__((ext_vector_type(4)))  _Float16 v4h;
typedef __attribute__((ext_vector_type(8)))  float    v8f;

static __device__ __forceinline__ v8f wmma_f16(v16h a, v16h b, v8f c) {
  // D = A(16x32) * B(32x16) + C(16x16 f32)
  return __builtin_amdgcn_wmma_f32_16x16x32_f16(false, a, false, b,
                                                (short)0, c, false, false);
}

// A-operand fragment: lane holds row (lane%16); per ISA 16-bit A layout the
// lane's 16 elements are K in [8*lh,8*lh+8) then [16+8*lh,16+8*lh+8), lh=lane/16.
static __device__ __forceinline__ v16h frag_a(const _Float16* rowp, int lh) {
  v16h f;
  ((v8h*)&f)[0] = *(const v8h*)(rowp + 8 * lh);
  ((v8h*)&f)[1] = *(const v8h*)(rowp + 16 + 8 * lh);
  return f;
}
// B-operand fragment: lane holds column (lane%16); elements K in
// [8*lh,8*lh+8) then [16+8*lh,16+8*lh+8), same order as frag_a.
static __device__ __forceinline__ v16h frag_b(const _Float16* colp, int lh) {
  v16h f;
  ((v8h*)&f)[0] = *(const v8h*)(colp + 8 * lh);
  ((v8h*)&f)[1] = *(const v8h*)(colp + 16 + 8 * lh);
  return f;
}

// CDNA5 async global->LDS DMA (ASYNCcnt-tracked), 16 bytes per lane.
// Low 32 bits of a generic pointer to __shared__ are the wave-relative LDS
// byte address (flat LDS aperture truncates to addr[31:0]).
static __device__ __forceinline__ void async_b128(void* lds, const void* g) {
  const unsigned int loff = (unsigned int)(unsigned long long)lds;
  asm volatile("global_load_async_to_lds_b128 %0, %1, off"
               :: "v"(loff), "v"(g) : "memory");
}
static __device__ __forceinline__ void wait_async0() {
  asm volatile("s_wait_asynccnt 0x0" ::: "memory");
}

// ------------------------------------------------------------------
// Kernel 1: fp32 -> f16 convert, optionally scaling a leading prefix.
// ------------------------------------------------------------------
__global__ void cvt_f32_to_f16(_Float16* __restrict__ dst,
                               const float* __restrict__ src,
                               int n4, int nScaled, float sc) {
  int i = blockIdx.x * blockDim.x + threadIdx.x;
  if (i >= n4) return;
  const float4 v = ((const float4*)src)[i];
  const float s = (i * 4 < nScaled) ? sc : 1.0f;
  v4h o;
  o[0] = (_Float16)(v.x * s); o[1] = (_Float16)(v.y * s);
  o[2] = (_Float16)(v.z * s); o[3] = (_Float16)(v.w * s);
  *(v4h*)(dst + (size_t)i * 4) = o;
}

// ------------------------------------------------------------------
// Kernel 2/4: tiled WMMA GEMM  out = A[Mx512] * W[ncolsx512]^T
//   128x128 block tile, 8 waves (2x4), each wave 64x32 (8 accs).
//   Async-to-LDS staging, double-buffered: one barrier per k-step,
//   next tile's DMA overlaps current tile's WMMAs.
//   mode 0: scatter f16 into Q/K/V [B,H,N,HD]   (ncols = 1536)
//   mode 1: f32 out + bias                      (ncols = 512)
// ------------------------------------------------------------------
__global__ __launch_bounds__(256)
void gemm_wmma(const _Float16* __restrict__ A, const _Float16* __restrict__ W,
               int mode,
               _Float16* __restrict__ q, _Float16* __restrict__ k,
               _Float16* __restrict__ v,
               float* __restrict__ outF, const float* __restrict__ bias) {
  __shared__ alignas(16) _Float16 Al[2][128 * 40];   // 32-wide K slab, +8 pad
  __shared__ alignas(16) _Float16 Bl[2][128 * 40];

  const int tid  = threadIdx.x;
  const int lane = tid & 31, wid = tid >> 5;
  const int lh = lane >> 4, l16 = lane & 15;
  const int wm = wid >> 2, wn = wid & 3;          // 2x4 wave grid
  const int row_base = blockIdx.x * 128;
  const int col_base = blockIdx.y * 128;

  const v8f vzero = {0.f, 0.f, 0.f, 0.f, 0.f, 0.f, 0.f, 0.f};
  v8f acc[4][2];
#pragma unroll
  for (int ms = 0; ms < 4; ++ms)
#pragma unroll
    for (int ns = 0; ns < 2; ++ns) acc[ms][ns] = vzero;

  const int sr   = tid >> 1;          // staging row 0..127
  const int sseg = (tid & 1) * 16;    // staging 16-f16 half
  const _Float16* gA = A + (size_t)(row_base + sr) * 512 + sseg;
  const _Float16* gB = W + (size_t)(col_base + sr) * 512 + sseg;

  // Preload first tile (k0 = 0) into buffer 0.
  async_b128(&Al[0][sr * 40 + sseg],     gA);
  async_b128(&Al[0][sr * 40 + sseg + 8], gA + 8);
  async_b128(&Bl[0][sr * 40 + sseg],     gB);
  async_b128(&Bl[0][sr * 40 + sseg + 8], gB + 8);

  int buf = 0;
  for (int k0 = 0; k0 < 512; k0 += 32) {
    wait_async0();       // own DMA into 'buf' complete
    __syncthreads();     // everyone's DMA complete; buf^1 free to overwrite
    if (k0 + 32 < 512) { // issue next tile while computing this one
      const int nb2 = buf ^ 1;
      async_b128(&Al[nb2][sr * 40 + sseg],     gA + k0 + 32);
      async_b128(&Al[nb2][sr * 40 + sseg + 8], gA + k0 + 40);
      async_b128(&Bl[nb2][sr * 40 + sseg],     gB + k0 + 32);
      async_b128(&Bl[nb2][sr * 40 + sseg + 8], gB + k0 + 40);
    }

    v16h af[4], bf[2];
#pragma unroll
    for (int ms = 0; ms < 4; ++ms)
      af[ms] = frag_a(&Al[buf][(wm * 64 + ms * 16 + l16) * 40], lh);
#pragma unroll
    for (int ns = 0; ns < 2; ++ns)
      bf[ns] = frag_b(&Bl[buf][(wn * 32 + ns * 16 + l16) * 40], lh);
#pragma unroll
    for (int ms = 0; ms < 4; ++ms)
#pragma unroll
      for (int ns = 0; ns < 2; ++ns)
        acc[ms][ns] = wmma_f16(af[ms], bf[ns], acc[ms][ns]);
    buf ^= 1;
  }

  // Epilogue: C/D layout row = r + 8*lh (+16*ms), col = lane%16 (+16*ns)
#pragma unroll
  for (int ms = 0; ms < 4; ++ms)
#pragma unroll
    for (int ns = 0; ns < 2; ++ns)
#pragma unroll
      for (int r = 0; r < 8; ++r) {
        const int rg = row_base + wm * 64 + ms * 16 + lh * 8 + r;
        const int cg = col_base + wn * 32 + ns * 16 + l16;
        const float val = acc[ms][ns][r];
        if (mode == 0) {
          const int sel = cg >> 9, h = (cg >> 7) & 3, d = cg & 127;
          const int bi = rg >> 10, n = rg & 1023;
          _Float16* dst = (sel == 0) ? q : (sel == 1) ? k : v;
          dst[(((size_t)bi * HH + h) * NN + n) * HD + d] = (_Float16)val;
        } else {
          outF[(size_t)rg * 512 + cg] = val + bias[cg];
        }
      }
}

// ------------------------------------------------------------------
// Kernel 3: fused CIM attention.
// Grid: (N/64, B). 8 waves = 4 heads x 2 row-chunks of 32 rows.
// Per m-block(32): S_h = Q_h K_h^T (WMMA) -> LDS; mix across heads with M;
// online softmax (lane-owns-row, no cross-lane reductions); P (f16) -> LDS;
// O = O*rescale + P·V_h (WMMA, V staged transposed).
// K tile is fetched with async-to-LDS DMA, overlapped with the VGPR-side
// V-transpose staging.
// ------------------------------------------------------------------
constexpr int KSTR = 136;   // 128 + 8 pad, f16
constexpr int VSTR = 40;    // 32 + 8 pad, f16 (transposed V)
constexpr int SSTR = 36;    // 32 + 4 pad, f32
constexpr int PSTR = 72;    // f16 view of the S region (36 f32 = 72 f16)
constexpr size_t KLDS_E = (size_t)4 * 32 * KSTR;
constexpr size_t VLDS_E = (size_t)4 * 128 * VSTR;
constexpr size_t SLDS_F = (size_t)8 * 32 * SSTR;
constexpr size_t ATTN_SMEM = (KLDS_E + VLDS_E) * 2 + SLDS_F * 4;  // 112640 B

__global__ __launch_bounds__(256)
void attn_cim(const _Float16* __restrict__ Q, const _Float16* __restrict__ Kt,
              const _Float16* __restrict__ V,
              const float* __restrict__ w_main, const float* __restrict__ w_rest,
              _Float16* __restrict__ AO) {
  extern __shared__ char smem[];
  _Float16* Klds = (_Float16*)smem;                 // [4][32][KSTR]
  _Float16* Vlds = Klds + KLDS_E;                   // [4][128][VSTR] (V^T)
  float*    Slds = (float*)(Vlds + VLDS_E);         // [8][32][SSTR]

  const int tid  = threadIdx.x;
  const int lane = tid & 31, wid = tid >> 5;
  const int lh = lane >> 4, l16 = lane & 15;
  const int head = wid >> 1, chunk = wid & 1;
  const int nb = blockIdx.x, b = blockIdx.y;
  const int row0 = nb * 64 + chunk * 32;

  // CIM mix-matrix row for output head i=head:
  //   M[i][j] = (j==i) ? w_main[i] : w_rest[i][j - (j>i)]
  float mco[4];
#pragma unroll
  for (int h = 0; h < 4; ++h)
    mco[h] = (h == head) ? w_main[head]
                         : w_rest[head * 3 + h - (h > head ? 1 : 0)];

  // Q fragments in registers: 32 rows x 128 dims = 2 m-subtiles x 4 k-steps.
  const _Float16* qbase = Q + ((size_t)(b * HH + head) * NN + row0) * HD;
  v16h qf[2][4];
#pragma unroll
  for (int ms = 0; ms < 2; ++ms)
#pragma unroll
    for (int ks = 0; ks < 4; ++ks)
      qf[ms][ks] = frag_a(qbase + (size_t)(ms * 16 + l16) * HD + ks * 32, lh);

  const v8f vzero = {0.f, 0.f, 0.f, 0.f, 0.f, 0.f, 0.f, 0.f};
  v8f o[2][8];
#pragma unroll
  for (int ms = 0; ms < 2; ++ms)
#pragma unroll
    for (int ds = 0; ds < 8; ++ds) o[ms][ds] = vzero;

  float mrow = -__builtin_huge_valf();
  float lrow = 0.0f;

  float*    mySl = Slds + (size_t)wid * 32 * SSTR;
  _Float16* myP  = (_Float16*)mySl;

  for (int mb = 0; mb < NN / 32; ++mb) {
    const int mg0 = mb * 32;
    __syncthreads();   // previous iteration's readers of Klds/Vlds are done
    // Async DMA: K tile [4][32][128] straight into LDS.
#pragma unroll
    for (int i = 0; i < 8; ++i) {
      const int idx = tid + i * 256;                 // [4][32][16 segs of 8]
      const int seg = idx & 15, m = (idx >> 4) & 31, h = idx >> 9;
      const _Float16* g = Kt + ((size_t)(b * HH + h) * NN + mg0 + m) * HD + seg * 8;
      async_b128(&Klds[(h * 32 + m) * KSTR + seg * 8], g);
    }
    // Meanwhile: stage V^T tile [4][128][32] through VGPRs (transpose).
#pragma unroll
    for (int i = 0; i < 8; ++i) {
      const int idx = tid + i * 256;
      const int seg = idx & 15, m = (idx >> 4) & 31, h = idx >> 9;
      const _Float16* g = V + ((size_t)(b * HH + h) * NN + mg0 + m) * HD + seg * 8;
      const v8h vv = *(const v8h*)g;
#pragma unroll
      for (int j = 0; j < 8; ++j)
        Vlds[(h * 128 + seg * 8 + j) * VSTR + m] = vv[j];
    }
    wait_async0();
    __syncthreads();

    // S = Q · K^T for (head, chunk): 32x32, K=128 in 4 steps.
    v8f s[2][2];
#pragma unroll
    for (int ms = 0; ms < 2; ++ms)
#pragma unroll
      for (int ns = 0; ns < 2; ++ns) s[ms][ns] = vzero;
#pragma unroll
    for (int ks = 0; ks < 4; ++ks) {
      v16h bf[2];
#pragma unroll
      for (int ns = 0; ns < 2; ++ns)
        bf[ns] = frag_b(&Klds[(head * 32 + ns * 16 + l16) * KSTR + ks * 32], lh);
#pragma unroll
      for (int ms = 0; ms < 2; ++ms)
#pragma unroll
        for (int ns = 0; ns < 2; ++ns)
          s[ms][ns] = wmma_f16(qf[ms][ks], bf[ns], s[ms][ns]);
    }
    // Publish S (f32) to this wave's region.
#pragma unroll
    for (int ms = 0; ms < 2; ++ms)
#pragma unroll
      for (int ns = 0; ns < 2; ++ns)
#pragma unroll
        for (int r = 0; r < 8; ++r)
          mySl[(ms * 16 + lh * 8 + r) * SSTR + ns * 16 + l16] = s[ms][ns][r];
    __syncthreads();

    // Cross-head mix; lane owns logits row (row0 + lane).
    float L[32];
#pragma unroll
    for (int c = 0; c < 32; ++c) L[c] = 0.0f;
#pragma unroll
    for (int h = 0; h < 4; ++h) {
      const float cf = mco[h];
      const float* srow = Slds + ((size_t)(h * 2 + chunk) * 32 + lane) * SSTR;
#pragma unroll
      for (int c = 0; c < 32; c += 4) {
        const float4 v4 = *(const float4*)(srow + c);
        L[c] += cf * v4.x; L[c + 1] += cf * v4.y;
        L[c + 2] += cf * v4.z; L[c + 3] += cf * v4.w;
      }
    }
    // Online softmax (per-lane row stats).
    float bmax = L[0];
#pragma unroll
    for (int c = 1; c < 32; ++c) bmax = fmaxf(bmax, L[c]);
    const float mnew = fmaxf(mrow, bmax);
    const float sc_old = __expf(mrow - mnew);       // exp(-inf)=0 first pass
    float bsum = 0.0f;
    float p[32];
#pragma unroll
    for (int c = 0; c < 32; ++c) { p[c] = __expf(L[c] - mnew); bsum += p[c]; }
    lrow = lrow * sc_old + bsum;
    mrow = mnew;

    __syncthreads();   // all waves done reading S before overwrite with P

    // P (f16) into own region; row stride 72 f16 (16B aligned).
#pragma unroll
    for (int c = 0; c < 32; c += 4) {
      v4h pk;
      pk[0] = (_Float16)p[c];     pk[1] = (_Float16)p[c + 1];
      pk[2] = (_Float16)p[c + 2]; pk[3] = (_Float16)p[c + 3];
      *(v4h*)&myP[lane * PSTR + c] = pk;
    }

    // Rescale O accumulators by exp(m_old - m_new) of their row.
#pragma unroll
    for (int ms = 0; ms < 2; ++ms)
#pragma unroll
      for (int r = 0; r < 8; ++r) {
        const float scr = __shfl(sc_old, ms * 16 + lh * 8 + r);
#pragma unroll
        for (int ds = 0; ds < 8; ++ds) o[ms][ds][r] *= scr;
      }

    // O += P · V  (K = 32, one WMMA step per output subtile).
    v16h pa[2];
#pragma unroll
    for (int ms = 0; ms < 2; ++ms)
      pa[ms] = frag_a(&myP[(ms * 16 + l16) * PSTR], lh);
#pragma unroll
    for (int ds = 0; ds < 8; ++ds) {
      const v16h vb = frag_b(&Vlds[(head * 128 + ds * 16 + l16) * VSTR], lh);
#pragma unroll
      for (int ms = 0; ms < 2; ++ms)
        o[ms][ds] = wmma_f16(pa[ms], vb, o[ms][ds]);
    }
  }

  // Epilogue: O /= row_sum, store f16 [B,N,C] with c = head*128 + d.
  const float inv = 1.0f / lrow;
#pragma unroll
  for (int ms = 0; ms < 2; ++ms)
#pragma unroll
    for (int r = 0; r < 8; ++r) {
      const float iv = __shfl(inv, ms * 16 + lh * 8 + r);
      const int rg = row0 + ms * 16 + lh * 8 + r;
#pragma unroll
      for (int ds = 0; ds < 8; ++ds) {
        const int cg = head * 128 + ds * 16 + l16;
        AO[((size_t)b * NN + rg) * CC + cg] = (_Float16)(o[ms][ds][r] * iv);
      }
    }
}

// ------------------------------------------------------------------
// Host launcher.  Workspace (f16 elems): X16 | WQ16 | WP16 | Q | K | V | AO
// ------------------------------------------------------------------
extern "C" void kernel_launch(void* const* d_in, const int* in_sizes, int n_in,
                              void* d_out, int out_size, void* d_ws, size_t ws_size,
                              hipStream_t stream) {
  (void)in_sizes; (void)n_in; (void)out_size; (void)ws_size;
  const float* x      = (const float*)d_in[0];
  const float* w_qkv  = (const float*)d_in[1];
  const float* w_proj = (const float*)d_in[2];
  const float* b_proj = (const float*)d_in[3];
  const float* w_main = (const float*)d_in[4];
  const float* w_rest = (const float*)d_in[5];
  float* out = (float*)d_out;

  _Float16* ws   = (_Float16*)d_ws;
  _Float16* X16  = ws;                 // 4,194,304
  _Float16* WQ16 = ws + 4194304;       //   786,432
  _Float16* WP16 = ws + 4980736;       //   262,144
  _Float16* Q16  = ws + 5242880;       // 4,194,304
  _Float16* K16  = ws + 9437184;       // 4,194,304
  _Float16* V16  = ws + 13631488;      // 4,194,304
  _Float16* AO16 = ws + 17825792;      // 4,194,304   (total ~44 MB)

  // 1) fp32 -> f16 (SCALE folded into the first 512 rows of w_qkv = Q weights)
  cvt_f32_to_f16<<<4096, 256, 0, stream>>>(X16, x, 1048576, 0, 1.0f);
  cvt_f32_to_f16<<<768, 256, 0, stream>>>(WQ16, w_qkv, 196608, 262144, QK_SCALE);
  cvt_f32_to_f16<<<256, 256, 0, stream>>>(WP16, w_proj, 65536, 0, 1.0f);

  // 2) QKV GEMM: [8192x512] x [1536x512]^T -> Q/K/V f16
  gemm_wmma<<<dim3(64, 12), 256, 0, stream>>>(X16, WQ16, 0,
                                              Q16, K16, V16, nullptr, nullptr);

  // 3) fused CIM attention -> AO16 [B,N,C] f16
  attn_cim<<<dim3(NN / 64, BB), 256, (unsigned)ATTN_SMEM, stream>>>(
      Q16, K16, V16, w_main, w_rest, AO16);

  // 4) projection GEMM + bias -> d_out f32
  gemm_wmma<<<dim3(64, 4), 256, 0, stream>>>(AO16, WP16, 1,
                                             nullptr, nullptr, nullptr,
                                             out, b_proj);
}
